// SelfAttentionLayer_55722905698551
// MI455X (gfx1250) — compile-verified, matrix operand order checked by reading, not run on hardware
//
#include <hip/hip_runtime.h>
#include <hip/hip_bf16.h>

// ---------------------------------------------------------------------------
// Self-attention (B=8, S=2048, D=256) for gfx1250 using bf16 WMMA +
// async global->LDS staging (ASYNCcnt) for the K/V stream.
// ---------------------------------------------------------------------------

typedef __bf16 bf16_t;
typedef __attribute__((ext_vector_type(16))) __bf16 v16bf;
typedef __attribute__((ext_vector_type(8)))  float  v8f;

#define B_N   8
#define S_LEN 2048
#define D_DIM 256

__device__ __forceinline__ bf16_t f2bf(float f) { return (bf16_t)f; }

// Load one 16-wide bf16 operand vector for WMMA A/B (16x32 tile, 16-bit layout):
//   element e in [0,7]  -> rowptr[koff + e]
//   element e in [8,15] -> rowptr[koff + 16 + (e-8)]
// Caller passes koff = k_base + 8*half. Works for global and LDS pointers.
template <typename P>
__device__ __forceinline__ v16bf load_tile16(const P* rowptr, int koff) {
  v16bf v;
  *(reinterpret_cast<uint4*>(&v))     = *(reinterpret_cast<const uint4*>(rowptr + koff));
  *(reinterpret_cast<uint4*>(&v) + 1) = *(reinterpret_cast<const uint4*>(rowptr + koff + 16));
  return v;
}

__device__ __forceinline__ v8f wmma_bf16(v16bf a, v16bf b, v8f c) {
  // D(16x16,f32) = A(16x32,bf16) * B(32x16,bf16) + C
  return __builtin_amdgcn_wmma_f32_16x16x32_bf16(
      /*neg_a=*/false, a, /*neg_b=*/false, b,
      /*c_mod=*/(short)0, c, /*reuse_a=*/false, /*reuse_b=*/false);
}

// Async global -> LDS copy of 16 bytes per lane (tracked by ASYNCcnt).
//   gbase : wave-uniform 64-bit global base (SGPR pair)
//   goff  : per-lane 32-bit byte offset (VGPR)
//   ldsa  : per-lane 32-bit LDS byte address (VGPR)
__device__ __forceinline__ void async_copy_b128(uint64_t gbase, uint32_t goff,
                                                uint32_t ldsa) {
  asm volatile("global_load_async_to_lds_b128 %0, %1, %2 offset:0"
               :: "v"(ldsa), "v"(goff), "s"(gbase)
               : "memory");
}

__device__ __forceinline__ void wait_asynccnt0() {
  asm volatile("s_wait_asynccnt 0x0" ::: "memory");
}

__device__ __forceinline__ uint32_t lds_addr(const void* p) {
  // generic shared pointer: low 32 bits are the LDS byte offset
  return (uint32_t)(uintptr_t)p;
}

// ---------------------------------------------------------------------------
// Kernel 0: fp32 -> bf16 conversion of activations + weights, zero output acc.
// ---------------------------------------------------------------------------
__global__ void prep_kernel(const float* __restrict__ nodes,
                            const float* __restrict__ Wq,
                            const float* __restrict__ Wk,
                            const float* __restrict__ Wv,
                            bf16_t* __restrict__ Xb,
                            bf16_t* __restrict__ Wqb,
                            bf16_t* __restrict__ Wkb,
                            bf16_t* __restrict__ Wvb,
                            float* __restrict__ out) {
  const int tid    = blockIdx.x * blockDim.x + threadIdx.x;
  const int stride = gridDim.x * blockDim.x;
  const int ntot = B_N * S_LEN * D_DIM;
  for (int i = tid; i < ntot; i += stride) Xb[i] = f2bf(nodes[i]);
  const int wtot = D_DIM * D_DIM;
  for (int i = tid; i < wtot; i += stride) {
    Wqb[i] = f2bf(Wq[i]);
    Wkb[i] = f2bf(Wk[i]);
    Wvb[i] = f2bf(Wv[i]);
  }
  if (tid < B_N * D_DIM) out[tid] = 0.0f;
}

// ---------------------------------------------------------------------------
// Kernel 1: QKV projection.  X[16384,256](bf16) @ W^T + b.
//   Q, K stored row-major bf16 [b][s][d]; V stored transposed bf16 [b][d][s].
//   grid = (128, 12): x = 128-row block, y = matrix(0..2) * 4 + colgroup(64).
// ---------------------------------------------------------------------------
__global__ __launch_bounds__(256)
void qkv_kernel(const bf16_t* __restrict__ X,
                const bf16_t* __restrict__ Wqb,
                const bf16_t* __restrict__ Wkb,
                const bf16_t* __restrict__ Wvb,
                const float* __restrict__ bq,
                const float* __restrict__ bk,
                const float* __restrict__ bv,
                bf16_t* __restrict__ Q,
                bf16_t* __restrict__ K,
                bf16_t* __restrict__ Vt) {
  const int lane = threadIdx.x & 31;
  const int wave = threadIdx.x >> 5;
  const int half = lane >> 4;
  const int l16  = lane & 15;

  const int mat  = blockIdx.y >> 2;          // 0=Q 1=K 2=V
  const int cg   = blockIdx.y & 3;           // 64-col group
  const int row0 = (blockIdx.x * 8 + wave) * 16;
  const int col0 = cg * 64;

  const bf16_t* W    = (mat == 0) ? Wqb : (mat == 1) ? Wkb : Wvb;
  const float*  bias = (mat == 0) ? bq  : (mat == 1) ? bk  : bv;

  const bf16_t* arow = X + (size_t)(row0 + l16) * D_DIM;

  v8f acc[4] = {};
  for (int c = 0; c < 8; ++c) {
    const int koff = c * 32 + half * 8;
    v16bf a = load_tile16(arow, koff);
    #pragma unroll
    for (int t = 0; t < 4; ++t) {
      const bf16_t* brow = W + (size_t)(col0 + t * 16 + l16) * D_DIM;
      v16bf bmat = load_tile16(brow, koff);
      acc[t] = wmma_bf16(a, bmat, acc[t]);
    }
  }

  if (mat < 2) {
    bf16_t* out = (mat == 0) ? Q : K;
    #pragma unroll
    for (int t = 0; t < 4; ++t) {
      const int   col = col0 + t * 16 + l16;
      const float bb  = bias[col];
      #pragma unroll
      for (int r = 0; r < 8; ++r) {
        const int m = r + half * 8;
        out[(size_t)(row0 + m) * D_DIM + col] = f2bf(acc[t][r] + bb);
      }
    }
  } else {
    #pragma unroll
    for (int t = 0; t < 4; ++t) {
      const int   col = col0 + t * 16 + l16;              // d
      const float bb  = bias[col];
      #pragma unroll
      for (int r = 0; r < 8; ++r) {
        const int g = row0 + r + half * 8;                // global token row
        const int b = g >> 11;
        const int s = g & 2047;
        Vt[((size_t)b * D_DIM + col) * S_LEN + s] = f2bf(acc[t][r] + bb);
      }
    }
  }
}

// ---------------------------------------------------------------------------
// Kernel 2: flash-attention, 32-key blocks staged into LDS with async copies
// (double-buffered), one 16-query tile per wave; 8 waves / workgroup share
// each staged K/V block.  grid = (B * 16), block = 256.
// ---------------------------------------------------------------------------
__global__ __launch_bounds__(256)
void attn_kernel(const bf16_t* __restrict__ Q,
                 const bf16_t* __restrict__ K,
                 const bf16_t* __restrict__ Vt,
                 const int* __restrict__ mask,
                 float* __restrict__ out) {
  __shared__ alignas(16) bf16_t kbuf[2][32 * D_DIM];   // 16 KB each: K rows
  __shared__ alignas(16) bf16_t vbuf[2][D_DIM * 32];   // 16 KB each: Vt rows
  __shared__ alignas(16) bf16_t pbuf[8][16 * 32];      // per-wave P staging

  const int tid  = threadIdx.x;
  const int lane = tid & 31;
  const int wave = tid >> 5;
  const int half = lane >> 4;
  const int l16  = lane & 15;

  const int blocks_per_batch = (S_LEN / 16) / 8;           // 16
  const int b  = blockIdx.x / blocks_per_batch;
  const int qt = (blockIdx.x % blocks_per_batch) * 8 + wave;
  const int q0 = qt * 16;

  const bf16_t* Qb = Q  + (size_t)b * S_LEN * D_DIM;
  const bf16_t* Kb = K  + (size_t)b * S_LEN * D_DIM;
  const bf16_t* Vb = Vt + (size_t)b * D_DIM * S_LEN;
  const int*    mb = mask + b * S_LEN;

  // ---- async staging of one 32-key block into buffer `buf`
  auto prefetch = [&](int blk, int buf) {
    // K block: contiguous 16 KB at Kb + blk*D_DIM
    {
      const uint64_t gbase = (uint64_t)(uintptr_t)(Kb + (size_t)blk * D_DIM);
      const uint32_t lbase = lds_addr(&kbuf[buf][0]);
      #pragma unroll
      for (int j = 0; j < 4; ++j) {
        const uint32_t off = (uint32_t)(tid + j * 256) * 16u;   // bytes
        async_copy_b128(gbase, off, lbase + off);
      }
    }
    // V block: 256 rows (d) x 64 B, global row stride S*2 bytes; row = tid
    {
      const uint64_t gbase = (uint64_t)(uintptr_t)(Vb + blk);
      const uint32_t lbase = lds_addr(&vbuf[buf][0]);
      #pragma unroll
      for (int j = 0; j < 4; ++j) {
        const uint32_t goff = (uint32_t)tid * (S_LEN * 2u) + (uint32_t)j * 16u;
        const uint32_t loff = (uint32_t)tid * 64u + (uint32_t)j * 16u;
        async_copy_b128(gbase, goff, lbase + loff);
      }
    }
  };

  // ---- Q tile as 8 persistent A operands (K = 0..255 in chunks of 32)
  v16bf qa[8];
  const bf16_t* qrow = Qb + (size_t)(q0 + l16) * D_DIM;
  #pragma unroll
  for (int c = 0; c < 8; ++c) qa[c] = load_tile16(qrow, c * 32 + half * 8);

  v8f o[16];
  #pragma unroll
  for (int t = 0; t < 16; ++t) o[t] = (v8f){};
  float Mr[8], Lr[8];
  #pragma unroll
  for (int r = 0; r < 8; ++r) { Mr[r] = -1e30f; Lr[r] = 0.0f; }

  const float scale = 0.0625f;   // 1/sqrt(256)
  const float NEG   = -1e30f;
  bf16_t* pw = &pbuf[wave][0];

  prefetch(0, 0);

  const int NBLK = S_LEN / 32;   // 64
  for (int i = 0; i < NBLK; ++i) {
    const int blk = i * 32;
    const int cur = i & 1;

    wait_asynccnt0();            // buffer `cur` fully landed in LDS
    __syncthreads();
    if (i + 1 < NBLK) prefetch(blk + 32, cur ^ 1);

    // ---- scores: S[16x32] = Q(16x256) . K^T from LDS
    v8f s0 = (v8f){}, s1 = (v8f){};
    const bf16_t* k0row = &kbuf[cur][l16 * D_DIM];
    const bf16_t* k1row = k0row + 16 * D_DIM;
    #pragma unroll
    for (int c = 0; c < 8; ++c) {
      const int koff = c * 32 + half * 8;
      v16bf kb0 = load_tile16(k0row, koff);
      s0 = wmma_bf16(qa[c], kb0, s0);
      v16bf kb1 = load_tile16(k1row, koff);
      s1 = wmma_bf16(qa[c], kb1, s1);
    }

    // ---- scale + key-column mask
    const int mv0 = mb[blk + l16];
    const int mv1 = mb[blk + 16 + l16];
    #pragma unroll
    for (int r = 0; r < 8; ++r) {
      s0[r] = mv0 ? s0[r] * scale : NEG;
      s1[r] = mv1 ? s1[r] * scale : NEG;
    }

    // ---- online softmax (row m = r + 8*half spans a 16-lane half-wave)
    float bmax[8];
    #pragma unroll
    for (int r = 0; r < 8; ++r) bmax[r] = fmaxf(s0[r], s1[r]);
    #pragma unroll
    for (int off = 8; off >= 1; off >>= 1)
      #pragma unroll
      for (int r = 0; r < 8; ++r)
        bmax[r] = fmaxf(bmax[r], __shfl_xor(bmax[r], off, 32));

    float alpha[8];
    #pragma unroll
    for (int r = 0; r < 8; ++r) {
      const float mn = fmaxf(Mr[r], bmax[r]);
      alpha[r] = __expf(Mr[r] - mn);
      Mr[r] = mn;
    }

    float rsum[8];
    #pragma unroll
    for (int r = 0; r < 8; ++r) {
      s0[r] = __expf(s0[r] - Mr[r]);
      s1[r] = __expf(s1[r] - Mr[r]);
      rsum[r] = s0[r] + s1[r];
    }
    #pragma unroll
    for (int off = 8; off >= 1; off >>= 1)
      #pragma unroll
      for (int r = 0; r < 8; ++r)
        rsum[r] += __shfl_xor(rsum[r], off, 32);
    #pragma unroll
    for (int r = 0; r < 8; ++r) Lr[r] = Lr[r] * alpha[r] + rsum[r];

    #pragma unroll
    for (int t = 0; t < 16; ++t)
      #pragma unroll
      for (int r = 0; r < 8; ++r) o[t][r] *= alpha[r];

    // ---- P (C-layout) -> LDS row-major [16][32] bf16, reload as A operand
    #pragma unroll
    for (int r = 0; r < 8; ++r) {
      const int m = r + half * 8;
      pw[m * 32 + l16]      = f2bf(s0[r]);
      pw[m * 32 + 16 + l16] = f2bf(s1[r]);
    }
    const bf16_t* prow = pw + l16 * 32;
    v16bf pa = load_tile16(prow, half * 8);

    // ---- O(16x256) += P(16x32) . V(32x256) from LDS
    #pragma unroll
    for (int t = 0; t < 16; ++t) {
      const bf16_t* vrow = &vbuf[cur][(t * 16 + l16) * 32];
      v16bf vtile = load_tile16(vrow, half * 8);
      o[t] = wmma_bf16(pa, vtile, o[t]);
    }

    __syncthreads();   // all waves done reading buffer `cur`
  }

  // ---- finalize: out[b][d] += sum_rows O / (L * S)
  const float invS = 1.0f / (float)S_LEN;
  #pragma unroll
  for (int r = 0; r < 8; ++r) {
    const float w = invS / Lr[r];
    #pragma unroll
    for (int t = 0; t < 16; ++t)
      atomicAdd(&out[b * D_DIM + t * 16 + l16], o[t][r] * w);
  }
}

// ---------------------------------------------------------------------------
// Launch
// ---------------------------------------------------------------------------
extern "C" void kernel_launch(void* const* d_in, const int* in_sizes, int n_in,
                              void* d_out, int out_size, void* d_ws, size_t ws_size,
                              hipStream_t stream) {
  const float* nodes = (const float*)d_in[0];
  const int*   mask  = (const int*)d_in[1];
  const float* Wq    = (const float*)d_in[2];
  const float* bq    = (const float*)d_in[3];
  const float* Wk    = (const float*)d_in[4];
  const float* bk    = (const float*)d_in[5];
  const float* Wv    = (const float*)d_in[6];
  const float* bv    = (const float*)d_in[7];
  float* out = (float*)d_out;

  char* ws = (char*)d_ws;
  const size_t actBytes = (size_t)B_N * S_LEN * D_DIM * sizeof(bf16_t); // 8 MiB
  const size_t wBytes   = (size_t)D_DIM * D_DIM * sizeof(bf16_t);       // 128 KiB
  bf16_t* Xb  = (bf16_t*)(ws);                 ws += actBytes;
  bf16_t* Qb  = (bf16_t*)(ws);                 ws += actBytes;
  bf16_t* Kb  = (bf16_t*)(ws);                 ws += actBytes;
  bf16_t* Vt  = (bf16_t*)(ws);                 ws += actBytes;
  bf16_t* Wqb = (bf16_t*)(ws);                 ws += wBytes;
  bf16_t* Wkb = (bf16_t*)(ws);                 ws += wBytes;
  bf16_t* Wvb = (bf16_t*)(ws);                 ws += wBytes;

  prep_kernel<<<4096, 256, 0, stream>>>(nodes, Wq, Wk, Wv, Xb, Wqb, Wkb, Wvb, out);

  dim3 gq(128, 12, 1);             // 128 row-blocks x (3 matrices * 4 colgroups)
  qkv_kernel<<<gq, 256, 0, stream>>>(Xb, Wqb, Wkb, Wvb, bq, bk, bv, Qb, Kb, Vt);

  attn_kernel<<<B_N * 16, 256, 0, stream>>>(Qb, Kb, Vt, mask, out);
}
